// RelationNet_43439299232227
// MI455X (gfx1250) — matrix operand order repaired and checked by reading, not compiled, not run on hardware
//
#include <hip/hip_runtime.h>

typedef _Float16 f16;
typedef __attribute__((ext_vector_type(16))) _Float16 v16h;
typedef __attribute__((ext_vector_type(8)))  float    v8f;
typedef __attribute__((ext_vector_type(4)))  float    v4f;
typedef unsigned long long u64;

#define BS 128
#define SL 200
#define M_ROWS (BS * SL)        // 25600
#define K1 3072
#define N1DIM 2048
#define K2 2048
#define N2DIM 1024
#define CD 1024
#define PD 768
#define SD 1024
#define FC 512
#define ZDIM (PD + N2DIM + SD)  // 2816
#define NHOP 34
#define KSTEPS1 (K1 / 32)       // 96
#define KSTEPS2 (K2 / 32)       // 64
#define NT1 (N1DIM / 16)        // 128 column tiles
#define NT2 (N2DIM / 16)        // 64 column tiles
#define BM 128                  // rows per workgroup (8 M-tiles)
#define BN 256                  // cols per workgroup (16 N-tiles)

__device__ __forceinline__ float gelu_tanh(float x) {
    float x3 = x * x * x;
    return 0.5f * x * (1.0f + tanhf(0.7978845608028654f * (x + 0.044715f * x3)));
}

union H4 { f16 h[4]; u64 v; };

// ---- pack weights f32[K,N] -> f16 fragment-native chunks ----
// chunk c = (step*NT + nt)*32 + lane  holds  W[step*32 + k(h,s), nt*16 + lane%16], h=lane/16, k = 8h+s (s<8), 16+8h+s-8 (s>=8)
__global__ __launch_bounds__(256) void pack_w_f16(const float* __restrict__ W,
                                                  f16* __restrict__ Wp,
                                                  int N, int ntlog2, int nchunks) {
    int c = blockIdx.x * 256 + threadIdx.x;
    if (c >= nchunks) return;
    const int lane = c & 31;
    const int tmp  = c >> 5;
    const int nt   = tmp & ((1 << ntlog2) - 1);
    const int step = tmp >> ntlog2;
    const float* src = W + (size_t)(step * 32 + (lane >> 4) * 8) * N + nt * 16 + (lane & 15);
    f16* dst = Wp + (size_t)c * 16;
    H4 o;
#pragma unroll
    for (int q = 0; q < 4; ++q) {
#pragma unroll
        for (int e = 0; e < 4; ++e) {
            const int s = q * 4 + e;
            const int kk = (s < 8) ? s : (s + 8);
            o.h[e] = (f16)src[(size_t)kk * N];
        }
        ((u64*)dst)[q] = o.v;
    }
}

// ---------------- fused gather/concat: build X[25600, 3072] f16 ----------------
__global__ __launch_bounds__(256) void gather_kernel(const int* __restrict__ qa_ids,
                                                     const int* __restrict__ rel_ids,
                                                     const float* __restrict__ concept_w,
                                                     const float* __restrict__ rel_w,
                                                     f16* __restrict__ X) {
    const int row = blockIdx.x;           // b*SL + s
    const int j   = threadIdx.x;          // float4 index 0..255
    const int qa0 = qa_ids[row * 2 + 0];
    const int qa1 = qa_ids[row * 2 + 1];
    const int rid = rel_ids[row];
    const v4f* c0 = (const v4f*)(concept_w + (size_t)qa0 * CD);
    const v4f* c1 = (const v4f*)(concept_w + (size_t)qa1 * CD);
    const bool twohop = rid >= NHOP;
    const int tw = (rid - NHOP) < 0 ? 0 : (rid - NHOP);
    const v4f* r1 = (const v4f*)(rel_w + (size_t)(twohop ? (tw / NHOP) : rid) * CD);
    const v4f* r2 = (const v4f*)(rel_w + (size_t)(tw % NHOP) * CD);
    f16* xr = X + (size_t)row * K1;

    v4f a = __builtin_nontemporal_load(&c0[j]);   // 400MB random table: keep out of L2
    H4 o;
    o.h[0] = (f16)a.x; o.h[1] = (f16)a.y; o.h[2] = (f16)a.z; o.h[3] = (f16)a.w;
    ((u64*)xr)[j] = o.v;

    v4f b = __builtin_nontemporal_load(&c1[j]);
    o.h[0] = (f16)b.x; o.h[1] = (f16)b.y; o.h[2] = (f16)b.z; o.h[3] = (f16)b.w;
    ((u64*)(xr + CD))[j] = o.v;

    v4f rv = r1[j];
    if (twohop) {
        v4f s = r2[j];
        rv.x *= s.x; rv.y *= s.y; rv.z *= s.z; rv.w *= s.w;
    }
    o.h[0] = (f16)rv.x; o.h[1] = (f16)rv.y; o.h[2] = (f16)rv.z; o.h[3] = (f16)rv.w;
    ((u64*)(xr + 2 * CD))[j] = o.v;
}

// ======== LDS-staged GEMM, 128x256 tile, 1024 threads (32 waves: 8 mi x 4 nq) ========
// Each unique A/B byte crosses L2 once per workgroup. acc = 4 fragments (32 VGPRs).

// ---------------- GEMM1: Y = X @ W0  (bias/LN/GELU done in ln_gelu_kernel) ----------
__global__ __launch_bounds__(1024) void gemm1_kernel(const f16* __restrict__ X,
                                                     const f16* __restrict__ Wp,
                                                     f16* __restrict__ Y) {
    __shared__ __align__(16) f16 As[2][BM * 32];        // 8 KB per buffer
    __shared__ __align__(16) f16 Bs[2][16 * 32 * 16];   // 16 KB per buffer

    const int t    = threadIdx.x;
    const int wid  = t >> 5;
    const int lane = t & 31;
    const int mi   = wid & 7;        // M-tile 0..7
    const int nq   = wid >> 3;       // N-quad 0..3 (4 tiles each)
    const int rwg  = blockIdx.x * BM;
    const int ng   = blockIdx.y;     // 256-col group

    v8f acc[4];
#pragma unroll
    for (int i = 0; i < 4; ++i) acc[i] = (v8f){0, 0, 0, 0, 0, 0, 0, 0};

    // A staging: 128 rows x 32 f16 = 1024 u64; one per thread
    const int srow = t >> 3;         // 0..127
    const int sd   = t & 7;          // u64 slot in row (8 per row)
    const f16* aptr = X + (size_t)(rwg + srow) * K1 + sd * 4;
    // B staging: 16 KB contiguous in packed layout = 2048 u64; two per thread
    const u64* bsrc = (const u64*)(Wp + (size_t)(ng * 16) * 512);
    const size_t bstep = (size_t)NT1 * 512 / 4;   // u64 per K-step of full packed W

    const int lrow = mi * 16 + (lane & 15);
    const int kh   = lane >> 4;

    u64 a0 = *(const u64*)aptr;
    u64 b0 = bsrc[t];
    u64 b1 = bsrc[t + 1024];

#pragma unroll 1
    for (int step = 0; step < KSTEPS1; ++step) {
        f16* abuf = As[step & 1];
        f16* bbuf = Bs[step & 1];
        ((u64*)abuf)[srow * 8 + sd] = a0;
        ((u64*)bbuf)[t]        = b0;
        ((u64*)bbuf)[t + 1024] = b1;
        __syncthreads();
        // branchless next-step prefetch (last iteration harmlessly re-loads)
        const int nx = (step + 1 < KSTEPS1) ? (step + 1) : (KSTEPS1 - 1);
        a0 = *(const u64*)(aptr + (size_t)nx * 32);
        b0 = bsrc[(size_t)nx * bstep + t];
        b1 = bsrc[(size_t)nx * bstep + t + 1024];

        union { v16h v; uint4 q[2]; } af;
        af.q[0] = *(const uint4*)(abuf + lrow * 32 + kh * 8);
        af.q[1] = *(const uint4*)(abuf + lrow * 32 + 16 + kh * 8);

        union { v16h v; uint4 q[2]; } bf[4];
#pragma unroll
        for (int u = 0; u < 4; ++u) {
            const uint4* bp = (const uint4*)(bbuf + ((nq * 4 + u) * 32 + lane) * 16);
            bf[u].q[0] = bp[0];
            bf[u].q[1] = bp[1];
        }
#pragma unroll
        for (int u = 0; u < 4; ++u)
            acc[u] = __builtin_amdgcn_wmma_f32_16x16x32_f16(
                false, af.v, false, bf[u].v, (short)0, acc[u], false, false);
    }

    const int ncol = lane & 15;
    const int rloc = mi * 16 + (lane >> 4) * 8;
#pragma unroll
    for (int u = 0; u < 4; ++u) {
        const int n = ng * BN + (nq * 4 + u) * 16 + ncol;
#pragma unroll
        for (int i = 0; i < 8; ++i)
            Y[(size_t)(rwg + rloc + i) * N1DIM + n] = (f16)acc[u][i];
    }
}

// ---------------- bias + LayerNorm(2048) + GELU: Y -> H (f16) ----------------
__global__ __launch_bounds__(256) void ln_gelu_kernel(const f16* __restrict__ Y,
                                                      const float* __restrict__ bias,
                                                      const float* __restrict__ gam,
                                                      const float* __restrict__ bet,
                                                      f16* __restrict__ H) {
    __shared__ float red[256];
    const int r = blockIdx.x;
    const int t = threadIdx.x;
    const f16* yr = Y + (size_t)r * N1DIM;

    union { uint4 q; f16 h[8]; } in;
    in.q = *(const uint4*)(yr + t * 8);
    float v[8];
    float ps = 0.0f, pq = 0.0f;
#pragma unroll
    for (int i = 0; i < 8; ++i) {
        float x = (float)in.h[i] + bias[t * 8 + i];
        v[i] = x;
        ps += x;
        pq += x * x;
    }
    red[t] = ps;
    __syncthreads();
    for (int o = 128; o > 0; o >>= 1) { if (t < o) red[t] += red[t + o]; __syncthreads(); }
    const float mean = red[0] * (1.0f / (float)N1DIM);
    __syncthreads();
    red[t] = pq;
    __syncthreads();
    for (int o = 128; o > 0; o >>= 1) { if (t < o) red[t] += red[t + o]; __syncthreads(); }
    const float var  = red[0] * (1.0f / (float)N1DIM) - mean * mean;
    const float rstd = rsqrtf(var + 1e-5f);

    union { uint4 q; f16 h[8]; } outp;
#pragma unroll
    for (int i = 0; i < 8; ++i) {
        float x = (v[i] - mean) * rstd * gam[t * 8 + i] + bet[t * 8 + i];
        outp.h[i] = (f16)gelu_tanh(x);
    }
    *(uint4*)(H + (size_t)r * N1DIM + t * 8) = outp.q;
}

// ---------------- GEMM2: qars = GELU(H @ W1 + b1)  (f32) ----------------
__global__ __launch_bounds__(1024) void gemm2_kernel(const f16* __restrict__ Hin,
                                                     const f16* __restrict__ Wp,
                                                     const float* __restrict__ bias,
                                                     float* __restrict__ qars) {
    __shared__ __align__(16) f16 As[2][BM * 32];
    __shared__ __align__(16) f16 Bs[2][16 * 32 * 16];

    const int t    = threadIdx.x;
    const int wid  = t >> 5;
    const int lane = t & 31;
    const int mi   = wid & 7;
    const int nq   = wid >> 3;
    const int rwg  = blockIdx.x * BM;
    const int ng   = blockIdx.y;     // 0..3

    v8f acc[4];
#pragma unroll
    for (int i = 0; i < 4; ++i) acc[i] = (v8f){0, 0, 0, 0, 0, 0, 0, 0};

    const int srow = t >> 3;
    const int sd   = t & 7;
    const f16* aptr = Hin + (size_t)(rwg + srow) * K2 + sd * 4;
    const u64* bsrc = (const u64*)(Wp + (size_t)(ng * 16) * 512);
    const size_t bstep = (size_t)NT2 * 512 / 4;

    const int lrow = mi * 16 + (lane & 15);
    const int kh   = lane >> 4;

    u64 a0 = *(const u64*)aptr;
    u64 b0 = bsrc[t];
    u64 b1 = bsrc[t + 1024];

#pragma unroll 1
    for (int step = 0; step < KSTEPS2; ++step) {
        f16* abuf = As[step & 1];
        f16* bbuf = Bs[step & 1];
        ((u64*)abuf)[srow * 8 + sd] = a0;
        ((u64*)bbuf)[t]        = b0;
        ((u64*)bbuf)[t + 1024] = b1;
        __syncthreads();
        const int nx = (step + 1 < KSTEPS2) ? (step + 1) : (KSTEPS2 - 1);
        a0 = *(const u64*)(aptr + (size_t)nx * 32);
        b0 = bsrc[(size_t)nx * bstep + t];
        b1 = bsrc[(size_t)nx * bstep + t + 1024];

        union { v16h v; uint4 q[2]; } af;
        af.q[0] = *(const uint4*)(abuf + lrow * 32 + kh * 8);
        af.q[1] = *(const uint4*)(abuf + lrow * 32 + 16 + kh * 8);

        union { v16h v; uint4 q[2]; } bf[4];
#pragma unroll
        for (int u = 0; u < 4; ++u) {
            const uint4* bp = (const uint4*)(bbuf + ((nq * 4 + u) * 32 + lane) * 16);
            bf[u].q[0] = bp[0];
            bf[u].q[1] = bp[1];
        }
#pragma unroll
        for (int u = 0; u < 4; ++u)
            acc[u] = __builtin_amdgcn_wmma_f32_16x16x32_f16(
                false, af.v, false, bf[u].v, (short)0, acc[u], false, false);
    }

    const int ncol = lane & 15;
    const int rloc = mi * 16 + (lane >> 4) * 8;
#pragma unroll
    for (int u = 0; u < 4; ++u) {
        const int n = ng * BN + (nq * 4 + u) * 16 + ncol;
        const float bv = bias[n];
#pragma unroll
        for (int i = 0; i < 8; ++i)
            qars[(size_t)(rwg + rloc + i) * N2DIM + n] = gelu_tanh(acc[u][i] + bv);
    }
}

// ---------------- masked mean pool over tuples (float4) ----------------
__global__ __launch_bounds__(256) void pool_kernel(const float* __restrict__ qars,
                                                   const int* __restrict__ num_tuples,
                                                   float* __restrict__ pooled) {
    const int b  = blockIdx.x;
    const int c4 = threadIdx.x;          // 256 float4 columns = 1024 floats
    int cnt = num_tuples[b];
    if (cnt < 1) cnt = 1;                // all-masked fix: keep tuple 0
    const float inv = 1.0f / (float)cnt;
    const v4f* base = (const v4f*)(qars + (size_t)b * SL * N2DIM);
    v4f s = {0.0f, 0.0f, 0.0f, 0.0f};
    for (int r = 0; r < cnt; ++r) {
        v4f v = base[(size_t)r * (N2DIM / 4) + c4];
        s.x += v.x; s.y += v.y; s.z += v.z; s.w += v.w;
    }
    v4f o = {s.x * inv, s.y * inv, s.z * inv, s.w * inv};
    ((v4f*)pooled)[b * (N2DIM / 4) + c4] = o;
}

// ---------------- output head: concat -> FC -> LN -> GELU -> logit ----------------
__global__ __launch_bounds__(256) void head_kernel(const float* __restrict__ path,
                                                   const float* __restrict__ pooled,
                                                   const float* __restrict__ sent,
                                                   const float* __restrict__ w0,
                                                   const float* __restrict__ b0,
                                                   const float* __restrict__ gam,
                                                   const float* __restrict__ bet,
                                                   const float* __restrict__ w1,
                                                   const float* __restrict__ b1,
                                                   float* __restrict__ out) {
    __shared__ float z[ZDIM];
    __shared__ float y[FC];
    __shared__ float red[256];
    const int b = blockIdx.x;
    const int t = threadIdx.x;

    for (int j = t; j < PD; j += 256) z[j] = path[b * PD + j];
    for (int j = t; j < N2DIM; j += 256) z[PD + j] = pooled[b * N2DIM + j];
    for (int j = t; j < SD; j += 256) z[PD + N2DIM + j] = sent[b * SD + j];
    __syncthreads();

#pragma unroll
    for (int jj = 0; jj < 2; ++jj) {
        const int j = t * 2 + jj;
        float s = b0[j];
        for (int k = 0; k < ZDIM; ++k) s += z[k] * w0[(size_t)k * FC + j];
        y[j] = s;
    }
    __syncthreads();

    float ps = 0.0f, pq = 0.0f;
    for (int j = t; j < FC; j += 256) { float v = y[j]; ps += v; pq += v * v; }
    red[t] = ps;
    __syncthreads();
    for (int o = 128; o > 0; o >>= 1) { if (t < o) red[t] += red[t + o]; __syncthreads(); }
    const float mean = red[0] * (1.0f / (float)FC);
    __syncthreads();
    red[t] = pq;
    __syncthreads();
    for (int o = 128; o > 0; o >>= 1) { if (t < o) red[t] += red[t + o]; __syncthreads(); }
    const float var = red[0] * (1.0f / (float)FC) - mean * mean;
    const float rstd = rsqrtf(var + 1e-5f);
    __syncthreads();

    float acc = 0.0f;
    for (int j = t; j < FC; j += 256) {
        float v = (y[j] - mean) * rstd * gam[j] + bet[j];
        acc += gelu_tanh(v) * w1[j];
    }
    red[t] = acc;
    __syncthreads();
    for (int o = 128; o > 0; o >>= 1) { if (t < o) red[t] += red[t + o]; __syncthreads(); }
    if (t == 0) out[b] = red[0] + b1[0];
}

// ---------------- workspace layout (bytes) ----------------
#define OFF_W0   ((size_t)0)
#define SZ_W0    ((size_t)K1 * N1DIM * 2)                 // 12.6 MB (packed)
#define OFF_W1   (OFF_W0 + SZ_W0)
#define SZ_W1    ((size_t)K2 * N2DIM * 2)                 // 4.2 MB (packed)
#define OFF_X    (OFF_W1 + SZ_W1)
#define SZ_X     ((size_t)M_ROWS * K1 * 2)                // 157 MB (qars f32 overlays: 105 MB)
#define OFF_Y    (OFF_X + SZ_X)
#define SZ_Y     ((size_t)M_ROWS * N1DIM * 2)             // 105 MB (pre-LN activations)
#define OFF_H    (OFF_Y + SZ_Y)
#define SZ_H     ((size_t)M_ROWS * N1DIM * 2)             // 105 MB
#define OFF_POOL (OFF_H + SZ_H)

extern "C" void kernel_launch(void* const* d_in, const int* in_sizes, int n_in,
                              void* d_out, int out_size, void* d_ws, size_t ws_size,
                              hipStream_t stream) {
    const float* path       = (const float*)d_in[0];
    const float* sent       = (const float*)d_in[1];
    const int*   qa_ids     = (const int*)d_in[2];
    const int*   rel_ids    = (const int*)d_in[3];
    const int*   num_tuples = (const int*)d_in[4];
    const float* concept_w  = (const float*)d_in[5];
    const float* rel_w      = (const float*)d_in[6];
    const float* mlp_w0     = (const float*)d_in[7];
    const float* mlp_b0     = (const float*)d_in[8];
    const float* mlp_ln_g   = (const float*)d_in[9];
    const float* mlp_ln_b   = (const float*)d_in[10];
    const float* mlp_w1     = (const float*)d_in[11];
    const float* mlp_b1     = (const float*)d_in[12];
    const float* out_w0     = (const float*)d_in[13];
    const float* out_b0     = (const float*)d_in[14];
    const float* out_ln_g   = (const float*)d_in[15];
    const float* out_ln_b   = (const float*)d_in[16];
    const float* out_w1     = (const float*)d_in[17];
    const float* out_b1     = (const float*)d_in[18];
    float* out = (float*)d_out;

    char* ws = (char*)d_ws;
    f16*   W0p    = (f16*)(ws + OFF_W0);
    f16*   W1p    = (f16*)(ws + OFF_W1);
    f16*   X      = (f16*)(ws + OFF_X);
    float* qars   = (float*)(ws + OFF_X);   // overlays X (X dead after gemm1)
    f16*   Ybuf   = (f16*)(ws + OFF_Y);
    f16*   Hbuf   = (f16*)(ws + OFF_H);
    float* pooled = (float*)(ws + OFF_POOL);

    const int nchunks1 = KSTEPS1 * NT1 * 32;   // 393216
    const int nchunks2 = KSTEPS2 * NT2 * 32;   // 131072
    pack_w_f16<<<(nchunks1 + 255) / 256, 256, 0, stream>>>(mlp_w0, W0p, N1DIM, 7, nchunks1);
    pack_w_f16<<<(nchunks2 + 255) / 256, 256, 0, stream>>>(mlp_w1, W1p, N2DIM, 6, nchunks2);

    gather_kernel<<<M_ROWS, 256, 0, stream>>>(qa_ids, rel_ids, concept_w, rel_w, X);

    gemm1_kernel<<<dim3(M_ROWS / BM, N1DIM / BN), 1024, 0, stream>>>(X, W0p, Ybuf);

    ln_gelu_kernel<<<M_ROWS, 256, 0, stream>>>(Ybuf, mlp_b0, mlp_ln_g, mlp_ln_b, Hbuf);

    gemm2_kernel<<<dim3(M_ROWS / BM, N2DIM / BN), 1024, 0, stream>>>(Hbuf, W1p, mlp_b1, qars);

    pool_kernel<<<BS, 256, 0, stream>>>(qars, num_tuples, pooled);

    head_kernel<<<BS, 256, 0, stream>>>(path, pooled, sent, out_w0, out_b0,
                                        out_ln_g, out_ln_b, out_w1, out_b1, out);
}
